// VelocityHead_21784074125413
// MI455X (gfx1250) — compile-verified, matrix operand order checked by reading, not run on hardware
//
#include <hip/hip_runtime.h>

typedef __attribute__((ext_vector_type(16))) _Float16 v16h_t;
typedef __attribute__((ext_vector_type(8)))  _Float16 v8h_t;
typedef __attribute__((ext_vector_type(8)))  float    v8f_t;

#define B_    64
#define T_    4096
#define D_    256
#define NTOK  (B_ * T_)
#define PI_F  3.14159265358979323846f

// ---------------------------------------------------------------------------
// Pack a row-major fp32 weight matrix W[k][n] (N=256) into f16 WMMA-B layout:
// entry index = ((ks*16 + nt)*32 + lane)*16 + h
//   n = nt*16 + (lane&15)
//   k = ks*32 + (h&7) + (h>>3)*16 + ((lane>>4)*8)
// k >= Krows is zero-padded (handles K=272 -> 9 k-steps of 32, and any pad).
// ---------------------------------------------------------------------------
__global__ void pack_b_kernel(const float* __restrict__ W, _Float16* __restrict__ dst,
                              int Krows, int ksteps) {
  int idx = blockIdx.x * blockDim.x + threadIdx.x;
  int total = ksteps * 16 * 32 * 16;
  if (idx >= total) return;
  int h    = idx & 15;
  int lane = (idx >> 4) & 31;
  int nt   = (idx >> 9) & 15;
  int ks   = idx >> 13;
  int n = nt * 16 + (lane & 15);
  int k = ks * 32 + (h & 7) + ((h >> 3) * 16) + ((lane >> 4) * 8);
  float v = (k < Krows) ? W[k * D_ + n] : 0.0f;
  dst[idx] = (_Float16)v;
}

// ---------------------------------------------------------------------------
// Per-token raw kinematic features (recomputed in both scan passes; L0 hits).
// ---------------------------------------------------------------------------
__device__ inline void raw_feat(const float* __restrict__ pos, const float* __restrict__ ts,
                                int b, int t,
                                float& dPx, float& dPy, float& dt, float& sp, float& dn) {
  if (t == 0) {
    dPx = 0.0f; dPy = 0.0f; dt = 1.0f;
  } else {
    int i = (b * T_ + t) * 2;
    dPx = pos[i]     - pos[i - 2];
    dPy = pos[i + 1] - pos[i - 1];
    float dr = ts[b * T_ + t] - ts[b * T_ + t - 1];
    dr = (dr > 0.01f) ? dr : 1.0f;
    dt = fminf(fmaxf(dr, 0.1f), 1000.0f);
  }
  float dist = sqrtf(dPx * dPx + dPy * dPy) + 1e-8f;
  float dts  = fminf(fmaxf(dt, 0.1f), 1000.0f) * 0.1f;
  sp = fminf(fmaxf(dist / (dts + 1e-6f), 0.0f), 10.0f);
  float dir = atan2f(dPy + 1e-8f, dPx + 1e-8f);
  dn = fminf(fmaxf(dir * (1.0f / PI_F), -1.0f), 1.0f);
}

// ---------------------------------------------------------------------------
// EMA over T via affine scan: y_t = 0.3 x_t + 0.7 y_{t-1}, y_0 = x_0.
// 128 threads x 32 steps; per-chunk transform is y_out = A*y_in + Bc with
// constant A = 0.7^32 (thread 0: A = 0 because y_0 ignores carry).
// One block per batch sequence.
// ---------------------------------------------------------------------------
__global__ __launch_bounds__(128)
void features_kernel(const float* __restrict__ pos, const float* __restrict__ ts,
                     float* __restrict__ feats, float* __restrict__ out_mf) {
  const int b   = blockIdx.x;
  const int tid = threadIdx.x;
  const int CH  = T_ / 128;       // 32
  const int t0  = tid * CH;
  const float A32 = 1.1044272754402031e-05f;  // 0.7^32

  __shared__ float sB0[128], sB1[128], cin0[128], cin1[128];

  // pass 1: local transform with zero carry-in
  float ys = 0.0f, yd = 0.0f;
  for (int i = 0; i < CH; ++i) {
    int t = t0 + i;
    float dPx, dPy, dt, sp, dn;
    raw_feat(pos, ts, b, t, dPx, dPy, dt, sp, dn);
    if (t == 0) { ys = sp; yd = dn; }
    else        { ys = 0.3f * sp + 0.7f * ys; yd = 0.3f * dn + 0.7f * yd; }
  }
  sB0[tid] = ys; sB1[tid] = yd;
  __syncthreads();

  // carry propagation (128 tiny affine compositions, negligible cost)
  if (tid == 0) {
    float cs = 0.0f, cd = 0.0f;
    for (int i = 0; i < 128; ++i) {
      cin0[i] = cs; cin1[i] = cd;
      float A = (i == 0) ? 0.0f : A32;
      cs = A * cs + sB0[i];
      cd = A * cd + sB1[i];
    }
  }
  __syncthreads();

  // pass 2: with true carry, emit features
  ys = cin0[tid]; yd = cin1[tid];
  for (int i = 0; i < CH; ++i) {
    int t = t0 + i;
    float dPx, dPy, dt, sp, dn;
    raw_feat(pos, ts, b, t, dPx, dPy, dt, sp, dn);
    if (t == 0) { ys = sp; yd = dn; }
    else        { ys = 0.3f * sp + 0.7f * ys; yd = 0.3f * dn + 0.7f * yd; }
    float mf = (ys > 0.1f) ? 1.0f : 0.0f;
    int tok = b * T_ + t;
    float* f = feats + (size_t)tok * 6;
    f[0] = dPx; f[1] = dPy; f[2] = dt; f[3] = ys; f[4] = yd; f[5] = mf;
    out_mf[tok] = mf;
  }
}

// ---------------------------------------------------------------------------
// WMMA A-operand loader from LDS (row-major f16, row stride 296 halves).
// lane<16 : row m=lane,    k in {K0..K0+7, K0+16..K0+23}
// lane>=16: row m=lane-16, k in {K0+8..K0+15, K0+24..K0+31}
// Two contiguous 16B ds loads per lane.
// ---------------------------------------------------------------------------
#define RSA 296
__device__ inline v16h_t load_A(const _Float16* hA, int lane, int K0) {
  int m    = lane & 15;
  int koff = (lane >> 4) * 8;
  const _Float16* rp = hA + m * RSA + K0 + koff;
  v8h_t lo = *(const v8h_t*)rp;
  v8h_t hi = *(const v8h_t*)(rp + 16);
  return __builtin_shufflevector(lo, hi, 0,1,2,3,4,5,6,7,8,9,10,11,12,13,14,15);
}

// ---------------------------------------------------------------------------
// Fused MLP: 16 tokens/block, 4 waves; each wave owns 4 N-tiles (64 cols).
// layer1 (K=6) VALU -> LN+relu -> f16 LDS; layer2/layer3 via WMMA f16.
// ---------------------------------------------------------------------------
__global__ __launch_bounds__(128)
void mlp_kernel(const float* __restrict__ feats,
                const _Float16* __restrict__ w2p, const _Float16* __restrict__ owp,
                const float* __restrict__ enc_w1, const float* __restrict__ enc_b1,
                const float* __restrict__ ln1_g,  const float* __restrict__ ln1_b,
                const float* __restrict__ enc_b2, const float* __restrict__ ln2_g,
                const float* __restrict__ ln2_b,  const float* __restrict__ emb,
                const float* __restrict__ out_b,  const float* __restrict__ mov_w,
                const float* __restrict__ mov_b,
                float* __restrict__ out_vel, float* __restrict__ out_lg) {
  __shared__ float    featsL[16][6];
  __shared__ _Float16 hA[16][RSA];      // A operand: 256 act + 16 emb + 16 zero + pad
  __shared__ float    y2L[16][260];     // pre-LN activations / vel staging
  __shared__ float    redS[16][8];
  __shared__ float    redQ[16][8];
  __shared__ float    mv[16][2];

  const int tid  = threadIdx.x;
  const int lane = tid & 31;
  const int wave = tid >> 5;
  const int tok0 = blockIdx.x * 16;

  if (tid < 96) {
    int r = tid / 6, c = tid % 6;
    featsL[r][c] = feats[(size_t)(tok0 + r) * 6 + c];
  }
  __syncthreads();

  // direction embedding + zero pad into hA[:,256:288]
  for (int s = tid; s < 512; s += 128) {
    int r = s >> 5, c = s & 31;
    float val = 0.0f;
    if (c < 16) {
      float dirs = featsL[r][4];
      float mf   = featsL[r][5];
      int bi = (int)((dirs + 1.0f) * 4.0f);     // (dir+1)*pi / (2pi) * 8
      bi = bi < 0 ? 0 : (bi > 7 ? 7 : bi);
      bi *= (int)mf;
      val = emb[bi * 16 + c];
    }
    hA[r][256 + c] = (_Float16)val;
  }

  // ---- layer 1: feats(16x6) @ enc_w1(6x256) + b1  (VALU, K=6) ----
  {
    int r = tid >> 3, c0 = (tid & 7) * 32;
    for (int c = 0; c < 32; ++c) {
      int n = c0 + c;
      float a = enc_b1[n];
#pragma unroll
      for (int k = 0; k < 6; ++k) a += featsL[r][k] * enc_w1[k * D_ + n];
      y2L[r][n] = a;
    }
  }
  __syncthreads();

  // ---- LN1 + relu -> hA[:,0:256] (f16) ----
  {
    int r = tid >> 3, sub = tid & 7, c0 = sub * 32;
    float s = 0.0f, q = 0.0f;
    for (int c = 0; c < 32; ++c) { float v = y2L[r][c0 + c]; s += v; q += v * v; }
    redS[r][sub] = s; redQ[r][sub] = q;
    __syncthreads();
    if (sub == 0) {
      float S = 0.0f, Q = 0.0f;
      for (int i = 0; i < 8; ++i) { S += redS[r][i]; Q += redQ[r][i]; }
      float m   = S * (1.0f / 256.0f);
      float var = Q * (1.0f / 256.0f) - m * m;
      mv[r][0] = m; mv[r][1] = rsqrtf(var + 1e-5f);
    }
    __syncthreads();
    for (int c = 0; c < 32; ++c) {
      int n = c0 + c;
      float v = (y2L[r][n] - mv[r][0]) * mv[r][1] * ln1_g[n] + ln1_b[n];
      v = fmaxf(v, 0.0f);
      hA[r][n] = (_Float16)v;
    }
  }
  __syncthreads();

  v8f_t zero8 = {0.f, 0.f, 0.f, 0.f, 0.f, 0.f, 0.f, 0.f};
  v8f_t acc[4];

  // ---- layer 2: h(16x256) @ enc_w2(256x256), WMMA f16, 8 k-steps ----
#pragma unroll
  for (int q = 0; q < 4; ++q) acc[q] = zero8;
  for (int ks = 0; ks < 8; ++ks) {
    v16h_t a = load_A(&hA[0][0], lane, ks * 32);
#pragma unroll
    for (int q = 0; q < 4; ++q) {
      int nt = wave * 4 + q;
      const v16h_t bmat = *(const v16h_t*)(w2p + (size_t)(((ks * 16) + nt) * 32 + lane) * 16);
      acc[q] = __builtin_amdgcn_wmma_f32_16x16x32_f16(false, a, false, bmat,
                                                      (short)0, acc[q], false, false);
    }
  }
  {
    int col15 = lane & 15;
    int r0    = (lane >> 4) * 8;    // C layout: vgpr j -> row j or 8+j
#pragma unroll
    for (int q = 0; q < 4; ++q) {
      int col = (wave * 4 + q) * 16 + col15;
      float bias = enc_b2[col];
#pragma unroll
      for (int j = 0; j < 8; ++j) y2L[r0 + j][col] = acc[q][j] + bias;
    }
  }
  __syncthreads();

  // ---- LN2 -> hA[:,0:256] (no relu) ----
  {
    int r = tid >> 3, sub = tid & 7, c0 = sub * 32;
    float s = 0.0f, q = 0.0f;
    for (int c = 0; c < 32; ++c) { float v = y2L[r][c0 + c]; s += v; q += v * v; }
    redS[r][sub] = s; redQ[r][sub] = q;
    __syncthreads();
    if (sub == 0) {
      float S = 0.0f, Q = 0.0f;
      for (int i = 0; i < 8; ++i) { S += redS[r][i]; Q += redQ[r][i]; }
      float m   = S * (1.0f / 256.0f);
      float var = Q * (1.0f / 256.0f) - m * m;
      mv[r][0] = m; mv[r][1] = rsqrtf(var + 1e-5f);
    }
    __syncthreads();
    for (int c = 0; c < 32; ++c) {
      int n = c0 + c;
      float v = (y2L[r][n] - mv[r][0]) * mv[r][1] * ln2_g[n] + ln2_b[n];
      hA[r][n] = (_Float16)v;
    }
  }
  __syncthreads();

  // ---- layer 3: [base | dir_e | 0](16x288) @ out_w_packed, 9 k-steps ----
#pragma unroll
  for (int q = 0; q < 4; ++q) acc[q] = zero8;
  for (int ks = 0; ks < 9; ++ks) {
    v16h_t a = load_A(&hA[0][0], lane, ks * 32);
#pragma unroll
    for (int q = 0; q < 4; ++q) {
      int nt = wave * 4 + q;
      const v16h_t bmat = *(const v16h_t*)(owp + (size_t)(((ks * 16) + nt) * 32 + lane) * 16);
      acc[q] = __builtin_amdgcn_wmma_f32_16x16x32_f16(false, a, false, bmat,
                                                      (short)0, acc[q], false, false);
    }
  }
  {
    int col15 = lane & 15;
    int r0    = (lane >> 4) * 8;
#pragma unroll
    for (int q = 0; q < 4; ++q) {
      int col = (wave * 4 + q) * 16 + col15;
      float bias = out_b[col];
#pragma unroll
      for (int j = 0; j < 8; ++j) {
        float v = acc[q][j] + bias;
        int row = r0 + j;
        out_vel[(size_t)(tok0 + row) * D_ + col] = v;
        y2L[row][col] = v;
      }
    }
  }
  __syncthreads();

  // ---- mov_logits: vel(16x256) @ mov_w(256x2) + mov_b ----
  if (tid < 32) {
    int m = tid >> 1, c = tid & 1;
    float s = mov_b[c];
    for (int k = 0; k < D_; ++k) s += y2L[m][k] * mov_w[k * 2 + c];
    out_lg[(size_t)(tok0 + m) * 2 + c] = s;
  }
}

// ---------------------------------------------------------------------------
extern "C" void kernel_launch(void* const* d_in, const int* in_sizes, int n_in,
                              void* d_out, int out_size, void* d_ws, size_t ws_size,
                              hipStream_t stream) {
  (void)in_sizes; (void)n_in; (void)out_size; (void)ws_size;
  const float* sensor_pos = (const float*)d_in[0];
  const float* timestamps = (const float*)d_in[1];
  const float* enc_w1 = (const float*)d_in[2];
  const float* enc_b1 = (const float*)d_in[3];
  const float* ln1_g  = (const float*)d_in[4];
  const float* ln1_b  = (const float*)d_in[5];
  const float* enc_w2 = (const float*)d_in[6];
  const float* enc_b2 = (const float*)d_in[7];
  const float* ln2_g  = (const float*)d_in[8];
  const float* ln2_b  = (const float*)d_in[9];
  const float* emb    = (const float*)d_in[10];
  const float* out_w  = (const float*)d_in[11];
  const float* out_b  = (const float*)d_in[12];
  const float* mov_w  = (const float*)d_in[13];
  const float* mov_b  = (const float*)d_in[14];

  // workspace layout: packed W2 (128KB f16) | packed out_w (144KB f16) | feats (6MB f32)
  _Float16* w2p = (_Float16*)d_ws;
  _Float16* owp = w2p + 8 * 16 * 32 * 16;                    // 65536 halves
  float*    fts = (float*)((char*)d_ws + 131072 + 147456);   // 278528 B offset

  float* out_vel = (float*)d_out;
  float* out_mf  = out_vel + (size_t)NTOK * D_;
  float* out_lg  = out_mf + NTOK;

  pack_b_kernel<<<(8 * 16 * 32 * 16 + 255) / 256, 256, 0, stream>>>(enc_w2, w2p, 256, 8);
  pack_b_kernel<<<(9 * 16 * 32 * 16 + 255) / 256, 256, 0, stream>>>(out_w, owp, 272, 9);
  features_kernel<<<B_, 128, 0, stream>>>(sensor_pos, timestamps, fts, out_mf);
  mlp_kernel<<<NTOK / 16, 128, 0, stream>>>(fts, w2p, owp, enc_w1, enc_b1, ln1_g, ln1_b,
                                            enc_b2, ln2_g, ln2_b, emb, out_b, mov_w, mov_b,
                                            out_vel, out_lg);
}
